// HardBinary_6373731467797
// MI455X (gfx1250) — matrix operand order repaired and checked by reading, not compiled, not run on hardware
//
#include <hip/hip_runtime.h>
#include <stdint.h>

typedef __attribute__((ext_vector_type(16))) _Float16 v16h;
typedef __attribute__((ext_vector_type(8)))  float    v8f;

namespace {
constexpr int K_TOT = 8192;               // IN_CHN
constexpr int N_TOT = 8192;               // OUT_CHN
constexpr int M_TOT = 64;                 // BATCH
constexpr int GEMM_BLOCK = 128;           // 4 waves (wave32)
constexpr int GEMM_GRID  = N_TOT / (4 * 16);   // 128 blocks; one 16-wide N tile per wave
constexpr int OUT_ELEMS  = M_TOT * N_TOT; // 524288
constexpr int SPLIT_K    = 4;             // deterministic K split
constexpr int K_PER_SPLIT = K_TOT / SPLIT_K;
}

union AFrag { uint4 q[2]; v16h h; };
union BFrag { unsigned short u[16]; v16h h; };

// ---- x: f32 -> f16, packed, enabling b128 A-fragment loads ------------------
__global__ __launch_bounds__(256) void k_convert_x(const float* __restrict__ x,
                                                   _Float16* __restrict__ xh) {
  int i = (blockIdx.x * blockDim.x + threadIdx.x) * 4;   // grid covers exactly M*K
  float4 v = *reinterpret_cast<const float4*>(x + i);
  union { _Float16 h[4]; uint2 u; } p;
  p.h[0] = (_Float16)v.x; p.h[1] = (_Float16)v.y;
  p.h[2] = (_Float16)v.z; p.h[3] = (_Float16)v.w;
  *reinterpret_cast<uint2*>(xh + i) = p.u;
}

// Load the 4 A fragments (one per 16-row M tile) for absolute K offset kOff.
// Per-lane halves: K = kOff + hi*8 + {0..7}, then kOff + 16 + hi*8 + {0..7}.
template<bool USE_XH>
__device__ __forceinline__ void loadA(AFrag (&a)[4], const float* __restrict__ x,
                                      const _Float16* __restrict__ xh,
                                      int r, int hi, int kOff) {
  #pragma unroll
  for (int mt = 0; mt < 4; ++mt) {
    if constexpr (USE_XH) {
      const _Float16* xr = xh + (size_t)(mt * 16 + r) * K_TOT + kOff + hi * 8;
      a[mt].q[0] = *reinterpret_cast<const uint4*>(xr);
      a[mt].q[1] = *reinterpret_cast<const uint4*>(xr + 16);
    } else {
      const float* xr = x + (size_t)(mt * 16 + r) * K_TOT + kOff + hi * 8;
      float4 f0 = *reinterpret_cast<const float4*>(xr);
      float4 f1 = *reinterpret_cast<const float4*>(xr + 4);
      float4 f2 = *reinterpret_cast<const float4*>(xr + 16);
      float4 f3 = *reinterpret_cast<const float4*>(xr + 20);
      float fv[16] = {f0.x, f0.y, f0.z, f0.w, f1.x, f1.y, f1.z, f1.w,
                      f2.x, f2.y, f2.z, f2.w, f3.x, f3.y, f3.z, f3.w};
      #pragma unroll
      for (int e = 0; e < 16; ++e) a[mt].h[e] = (_Float16)fv[e];
    }
  }
}

// ---- fused kernel: outBuf = x @ sign(W)^T over a K slice + partial sum|W| ---
// One wave owns one 16-wide N tile for all M=64 (4 WMMA accumulator chains).
// blockIdx.y selects the K slice; W is read exactly once across the whole grid,
// each iteration lanes r, r+16 read 32 consecutive floats of W row n0+r.
// A fragments are software-pipelined one iteration ahead so no WMMA waits on a
// load issued in the same iteration.
template<bool USE_XH>
__global__ __launch_bounds__(GEMM_BLOCK) void k_binlinear(
    const float* __restrict__ x,
    const float* __restrict__ w,
    const _Float16* __restrict__ xh,
    float* __restrict__ outBuf,          // [gridDim.y][64][8192] (unscaled)
    float* __restrict__ absPartial,      // [gridDim.y * gridDim.x]
    int kLen)                            // K elements per blockIdx.y slice
{
  const int tid  = threadIdx.x;
  const int lane = tid & 31;
  const int r    = lane & 15;
  const int hi   = lane >> 4;
  const int wave = tid >> 5;
  const int n0   = (blockIdx.x * 4 + wave) * 16;
  const int kBase = blockIdx.y * kLen;

  v8f acc[4] = {};
  float abs0 = 0.0f, abs1 = 0.0f;        // two chains to halve serial FP latency

  const float* wcol = w + (size_t)(n0 + r) * K_TOT + kBase + hi * 8;  // lane = N row; K = hi*8 + {0..7}, 16 + hi*8 + {0..7}

  AFrag aCur[4];
  loadA<USE_XH>(aCur, x, xh, r, hi, kBase);          // prologue: first A tiles

  #pragma unroll 2
  for (int kc = 0; kc < kLen; kc += 32) {
    // ---- W values for this iteration (16 coalesced b32 streams) ------------
    float wv[16];
    #pragma unroll
    for (int j = 0; j < 16; ++j) wv[j] = wcol[kc + (j < 8 ? j : j + 8)];

    // ---- prefetch NEXT iteration's A fragments (clamped, always in-bounds) -
    AFrag aNxt[4];
    {
      int kn = kc + 32;
      if (kn >= kLen) kn = kc;                       // last iter: re-read current
      loadA<USE_XH>(aNxt, x, xh, r, hi, kBase + kn);
    }

    // ---- B fragment: lane holds column N=n0+r; element j <-> K=kBase+kc+hi*8+(j<8 ? j : j+8) --
    BFrag b;
    #pragma unroll
    for (int j = 0; j < 16; ++j) {
      if (j & 1) abs1 += fabsf(wv[j]); else abs0 += fabsf(wv[j]);
      unsigned short s16 =
          (unsigned short)(0x3C00u | ((__float_as_uint(wv[j]) >> 16) & 0x8000u));
      b.u[j] = (wv[j] != 0.0f) ? s16 : (unsigned short)0;  // sign(+-0)=0
    }

    // ---- 4 independent WMMA chains on the PREVIOUSLY loaded A --------------
    #pragma unroll
    for (int mt = 0; mt < 4; ++mt) {
      // 8 args: (neg_a, A, neg_b, B, c_mod, C, reuse_a, reuse_b)
      acc[mt] = __builtin_amdgcn_wmma_f32_16x16x32_f16(
          false, aCur[mt].h, false, b.h, (short)0, acc[mt], false, false);
    }

    // rotate double buffer (register renaming under unroll)
    #pragma unroll
    for (int mt = 0; mt < 4; ++mt) aCur[mt] = aNxt[mt];
  }

  // D layout: VGPR v -> M = hi*8 + v, N = n0 + r. Coalesced per-VGPR rows.
  float* outT = outBuf + (size_t)blockIdx.y * OUT_ELEMS;
  #pragma unroll
  for (int mt = 0; mt < 4; ++mt) {
    #pragma unroll
    for (int v = 0; v < 8; ++v) {
      int m = mt * 16 + hi * 8 + v;
      outT[(size_t)m * N_TOT + n0 + r] = acc[mt][v];
    }
  }

  // Deterministic block reduction of sum|W| (fixed order).
  __shared__ float red[GEMM_BLOCK];
  red[tid] = abs0 + abs1;
  __syncthreads();
  if (tid == 0) {
    float s = 0.0f;
    for (int i = 0; i < GEMM_BLOCK; ++i) s += red[i];
    absPartial[blockIdx.y * gridDim.x + blockIdx.x] = s;
  }
}

// ---- fold partials -> scale = mean(|W|), deterministic fixed-order tree -----
__global__ void k_reduce_scale(const float* __restrict__ p,
                               float* __restrict__ scale, int n) {
  __shared__ float red[32];
  const int tid = threadIdx.x;             // 32 threads
  const int per = n / 32;                  // n is a multiple of 32
  float s = 0.0f;
  for (int i = 0; i < per; ++i) s += p[tid * per + i];
  red[tid] = s;
  __syncthreads();
  if (tid == 0) {
    float t = 0.0f;
    for (int i = 0; i < 32; ++i) t += red[i];
    *scale = t * (1.0f / 67108864.0f);     // / (8192*8192), exact pow2
  }
}

// ---- out = scale * (sum of SPLIT_K partial planes), fixed order -------------
__global__ __launch_bounds__(256) void k_combine_scale(
    const float* __restrict__ parts, const float* __restrict__ scale,
    float* __restrict__ out) {
  int i = (blockIdx.x * blockDim.x + threadIdx.x) * 4;  // grid covers exactly M*N
  float s = *scale;
  float4 a = *reinterpret_cast<const float4*>(parts + i);
  #pragma unroll
  for (int t = 1; t < SPLIT_K; ++t) {
    float4 b = *reinterpret_cast<const float4*>(parts + (size_t)t * OUT_ELEMS + i);
    a.x += b.x; a.y += b.y; a.z += b.z; a.w += b.w;
  }
  a.x *= s; a.y *= s; a.z *= s; a.w *= s;
  *reinterpret_cast<float4*>(out + i) = a;
}

// ---- out *= scale (non-split fallback) --------------------------------------
__global__ __launch_bounds__(256) void k_scale_out(float* __restrict__ out,
                                                   const float* __restrict__ scale) {
  int i = (blockIdx.x * blockDim.x + threadIdx.x) * 4;
  float s = *scale;
  float4 v = *reinterpret_cast<float4*>(out + i);
  v.x *= s; v.y *= s; v.z *= s; v.w *= s;
  *reinterpret_cast<float4*>(out + i) = v;
}

extern "C" void kernel_launch(void* const* d_in, const int* in_sizes, int n_in,
                              void* d_out, int out_size, void* d_ws, size_t ws_size,
                              hipStream_t stream) {
  const float* x = (const float*)d_in[0];   // [64, 8192] f32
  const float* w = (const float*)d_in[1];   // [8192, 8192] f32
  float* out = (float*)d_out;               // [64, 8192] f32

  const size_t XH_BYTES    = (size_t)M_TOT * K_TOT * sizeof(_Float16);       // 1 MB
  const size_t PARTS_BYTES = (size_t)SPLIT_K * OUT_ELEMS * sizeof(float);    // 8 MB
  const size_t ABS_BYTES   = (size_t)SPLIT_K * GEMM_GRID * sizeof(float);    // 2 KB
  char* ws = (char*)d_ws;

  const int convGrid  = (M_TOT * K_TOT) / (256 * 4);   // 512
  const int elemGrid  = OUT_ELEMS / (256 * 4);         // 512

  if (ws_size >= XH_BYTES + PARTS_BYTES + ABS_BYTES + 64) {
    // Tier A: f16 x + deterministic split-K=4 (2048 waves, ~2 per SIMD32)
    _Float16* xh    = (_Float16*)ws;
    float* parts    = (float*)(ws + XH_BYTES);
    float* absParts = (float*)(ws + XH_BYTES + PARTS_BYTES);
    float* scale    = absParts + SPLIT_K * GEMM_GRID;
    k_convert_x<<<convGrid, 256, 0, stream>>>(x, xh);
    k_binlinear<true><<<dim3(GEMM_GRID, SPLIT_K), GEMM_BLOCK, 0, stream>>>(
        x, w, xh, parts, absParts, K_PER_SPLIT);
    k_reduce_scale<<<1, 32, 0, stream>>>(absParts, scale, SPLIT_K * GEMM_GRID);
    k_combine_scale<<<elemGrid, 256, 0, stream>>>(parts, scale, out);
  } else if (ws_size >= XH_BYTES + 512 + 16) {
    // Tier B: f16 x, no split
    _Float16* xh    = (_Float16*)ws;
    float* absParts = (float*)(ws + XH_BYTES);
    float* scale    = (float*)(ws + XH_BYTES + 512);
    k_convert_x<<<convGrid, 256, 0, stream>>>(x, xh);
    k_binlinear<true><<<dim3(GEMM_GRID, 1), GEMM_BLOCK, 0, stream>>>(
        x, w, xh, out, absParts, K_TOT);
    k_reduce_scale<<<1, 32, 0, stream>>>(absParts, scale, GEMM_GRID);
    k_scale_out<<<elemGrid, 256, 0, stream>>>(out, scale);
  } else {
    // Tier C: tiny workspace — convert x inline inside the GEMM
    float* absParts = (float*)ws;
    float* scale    = (float*)(ws + 512);
    k_binlinear<false><<<dim3(GEMM_GRID, 1), GEMM_BLOCK, 0, stream>>>(
        x, w, nullptr, out, absParts, K_TOT);
    k_reduce_scale<<<1, 32, 0, stream>>>(absParts, scale, GEMM_GRID);
    k_scale_out<<<elemGrid, 256, 0, stream>>>(out, scale);
  }
}
